// Coattn_29231547417132
// MI455X (gfx1250) — matrix operand order repaired and checked by reading, not compiled, not run on hardware
//
#include <hip/hip_runtime.h>
#include <hip/hip_bf16.h>
#include <math.h>

// ---------------- CDNA5 wave32 WMMA co-attention kernel (round 2) ----------------
// One workgroup (512 thr = 16 wave32) per batch element b.
// c/s streamed HBM->LDS with global_load_async_to_lds_b128 (ASYNCcnt),
// all GEMMs via v_wmma_f32_16x16x32_bf16, all B operands read contiguously.

typedef __bf16 bf16;
typedef __attribute__((ext_vector_type(16))) __bf16 v16bf;
typedef __attribute__((ext_vector_type(8)))  __bf16 v8bf;
typedef __attribute__((ext_vector_type(8)))  float  v8f;

#define BT   512           // threads per block
#define NW   16            // waves per block
#define PD   104           // pitch for [*,96]-K bf16 matrices (104=8*13 -> 16B-aligned rows)
#define PT   136           // pitch for [*,128]-K bf16 matrices (136=8*17)
// LDS element counts (bf16)
#define N_CD   (112*104)   // 11648 : cS / sS / WcS / WwS
#define N_WCWT (80*104)    // 8320  : W_cw transposed [l][d]
#define N_F    (112*136)   // 15232 : Fs (row major [t][n]) and FsT ([n][t]); R1/R2 same size
#define SMEM_ELEMS (4*N_CD + N_WCWT + 4*N_F)               // 115840 bf16
#define SMEM_BYTES (SMEM_ELEMS*2 + 4*112*4)                // 233472 B  (< 320KB/WGP)

#if __has_builtin(__builtin_amdgcn_tanhf)
#define TANH(x) __builtin_amdgcn_tanhf(x)
#elif __has_builtin(__builtin_amdgcn_tanh_f32)
#define TANH(x) __builtin_amdgcn_tanh_f32(x)
#else
#define TANH(x) tanhf(x)
#endif

union V16 { v16bf v; v8bf h[2]; };

__device__ __forceinline__ v16bf ldA(const bf16* base, int pitch, int row0, int k0) {
    int lane = threadIdx.x & 31;
    const bf16* p = base + (row0 + (lane & 15)) * pitch + k0 + ((lane >> 4) << 3);
    V16 u;
    u.h[0] = *(const v8bf*)p;        // K = k0+8h .. +7
    u.h[1] = *(const v8bf*)(p + 16); // K = k0+16+8h .. +7
    return u.v;
}

// B operand where the source holds B transposed: B[k][n] = S[n][k]  (contiguous per lane)
__device__ __forceinline__ v16bf ldBt(const bf16* base, int pitch, int n0, int k0) {
    int lane = threadIdx.x & 31;
    const bf16* p = base + (n0 + (lane & 15)) * pitch + k0 + ((lane >> 4) << 3);
    V16 u;
    u.h[0] = *(const v8bf*)p;        // K = k0+8h .. +7
    u.h[1] = *(const v8bf*)(p + 16); // K = k0+16+8h .. +7
    return u.v;
}

__device__ __forceinline__ v8f ldC(const bf16* base, int pitch, int row0, int n0) {
    int lane = threadIdx.x & 31;
    const bf16* p = base + (row0 + ((lane >> 4) << 3)) * pitch + n0 + (lane & 15);
    v8f c;
#pragma unroll
    for (int v = 0; v < 8; ++v) c[v] = (float)p[v * pitch];
    return c;
}

// D tile -> row-major [row][col] (per-lane column segment: strided 16-bit stores)
__device__ __forceinline__ void stD(bf16* dst, int pitch, int row0, int n0, v8f acc) {
    int lane = threadIdx.x & 31;
    bf16* p = dst + (row0 + ((lane >> 4) << 3)) * pitch + n0 + (lane & 15);
#pragma unroll
    for (int v = 0; v < 8; ++v) p[v * pitch] = (bf16)acc[v];
}

// D tile -> transposed target Dt[col][row] (per-lane contiguous: one b128 store)
__device__ __forceinline__ void stDT(bf16* dst, int pitch, int row0, int n0, v8f acc) {
    int lane = threadIdx.x & 31;
    bf16* p = dst + (n0 + (lane & 15)) * pitch + row0 + ((lane >> 4) << 3);
    v8bf r;
#pragma unroll
    for (int v = 0; v < 8; ++v) r[v] = (bf16)acc[v];
    *(v8bf*)p = r;
}

#define WMMA_BF16(a, b, c) \
    __builtin_amdgcn_wmma_f32_16x16x32_bf16(false, (a), false, (b), (short)0, (c), false, false)

__global__ __launch_bounds__(BT, 1) void coattn_kernel(
    const float* __restrict__ Cg,   // [B,100,80]
    const float* __restrict__ Sg,   // [B,100,80]
    const float* __restrict__ Wcw,  // [80,80]
    const float* __restrict__ Wc,   // [100,80]
    const float* __restrict__ Ww,   // [100,80]
    const float* __restrict__ whw,  // [100]
    const float* __restrict__ whc,  // [100]
    float* __restrict__ Out)        // [B,160]
{
    extern __shared__ __align__(16) char smem_raw[];
    bf16* cS   = (bf16*)smem_raw;          // [112][104]
    bf16* sS   = cS   + N_CD;
    bf16* WcS  = sS   + N_CD;
    bf16* WwS  = WcS  + N_CD;
    bf16* WcwT = WwS  + N_CD;              // [80][104]  W_cw^T  (row l, col d)
    bf16* Fs   = WcwT + N_WCWT;            // [112][136] F row-major (also early: f32 staging)
    bf16* FsT  = Fs   + N_F;               // [112][136] F transposed
    bf16* R1   = FsT  + N_F;               // P (pitch 104) then Cw (pitch 136)
    bf16* R2   = R1   + N_F;               // Sw (pitch 136)
    float* whwS = (float*)(R2 + N_F);      // [112]
    float* whcS = whwS + 112;
    float* awS  = whcS + 112;
    float* acS  = awS  + 112;

    const int tid  = threadIdx.x;
    const int wid  = tid >> 5;
    const int lane = tid & 31;
    const int b    = blockIdx.x;
    const float* cb = Cg + (size_t)b * 100 * 80;
    const float* sb = Sg + (size_t)b * 100 * 80;

    // ---- Phase 0a: async-stream c,s (32000 B each) into LDS f32 staging ----
    float* stageC = (float*)Fs;                  // 8000 f32
    float* stageS = (float*)Fs + 8000;           // 8000 f32 (spills into FsT; reclaimed later)
    const unsigned ldsBase = __builtin_amdgcn_groupstaticsize();
    const unsigned cOff = ldsBase + (unsigned)((char*)stageC - smem_raw);
    const unsigned sOff = ldsBase + (unsigned)((char*)stageS - smem_raw);
    for (int i = tid; i < 2000; i += BT) {       // 2000 x 16B = 32000 B
        asm volatile("global_load_async_to_lds_b128 %0, %1, off"
                     :: "v"(cOff + i * 16),
                        "v"((unsigned long long)(const void*)(cb + i * 4))
                     : "memory");
        asm volatile("global_load_async_to_lds_b128 %0, %1, off"
                     :: "v"(sOff + i * 16),
                        "v"((unsigned long long)(const void*)(sb + i * 4))
                     : "memory");
    }

    // ---- Phase 0b (overlapped): stage weights as bf16 (zero-fill pads) ----
    for (int idx = tid; idx < N_CD; idx += BT) {
        int r = idx / PD, c = idx % PD;
        bool ok = (r < 100) && (c < 80);
        WcS[idx] = ok ? (bf16)Wc[r * 80 + c] : (bf16)0.f;
        WwS[idx] = ok ? (bf16)Ww[r * 80 + c] : (bf16)0.f;
    }
    for (int idx = tid; idx < N_WCWT; idx += BT) {
        int l = idx / PD, d = idx % PD;          // transposed: row l, col d
        WcwT[idx] = (l < 80 && d < 80) ? (bf16)Wcw[d * 80 + l] : (bf16)0.f;
    }
    for (int idx = tid; idx < 112; idx += BT) {
        whwS[idx] = (idx < 100) ? whw[idx] : 0.f;
        whcS[idx] = (idx < 100) ? whc[idx] : 0.f;
        awS[idx] = 0.f; acS[idx] = 0.f;
    }

    asm volatile("s_wait_asynccnt 0" ::: "memory");
    __syncthreads();

    // ---- Phase 0c: convert staged f32 c,s -> bf16 (padded) ----
    for (int idx = tid; idx < N_CD; idx += BT) {
        int r = idx / PD, c = idx % PD;
        bool ok = (r < 100) && (c < 80);
        cS[idx] = ok ? (bf16)stageC[r * 80 + c] : (bf16)0.f;
        sS[idx] = ok ? (bf16)stageS[r * 80 + c] : (bf16)0.f;
    }
    __syncthreads();

    // ---- Phase 0d: zero Fs/FsT/R1/R2 (covers all K-pads; staging now dead) ----
    for (int idx = tid; idx < N_F; idx += BT) {
        Fs[idx] = (bf16)0.f; FsT[idx] = (bf16)0.f;
        R1[idx] = (bf16)0.f; R2[idx] = (bf16)0.f;
    }
    __syncthreads();

    // ---- Phase 1: P = c @ W_cw   [112 x 80], K=96 -> R1 (pitch 104) ----
    for (int q = wid; q < 7 * 5; q += NW) {
        int ti = q / 5, tj = q % 5;
        v8f acc = {};
#pragma unroll
        for (int kc = 0; kc < 3; ++kc) {
            v16bf a  = ldA(cS, PD, ti * 16, kc * 32);
            v16bf bb = ldBt(WcwT, PD, tj * 16, kc * 32);   // B[d][l] = WcwT[l][d]
            acc = WMMA_BF16(a, bb, acc);
        }
        stD(R1, PD, ti * 16, tj * 16, acc);
    }
    __syncthreads();

    // ---- Phase 2: F = tanh(P @ s^T)  [112 x 112], K=96 -> Fs + FsT ----
    for (int q = wid; q < 49; q += NW) {
        int ti = q / 7, tj = q % 7;
        v8f acc = {};
#pragma unroll
        for (int kc = 0; kc < 3; ++kc) {
            v16bf a  = ldA(R1, PD, ti * 16, kc * 32);
            v16bf bb = ldBt(sS, PD, tj * 16, kc * 32);     // B = s^T
            acc = WMMA_BF16(a, bb, acc);
        }
#pragma unroll
        for (int v = 0; v < 8; ++v) acc[v] = TANH(acc[v]);
        stD (Fs,  PT, ti * 16, tj * 16, acc);
        stDT(FsT, PT, ti * 16, tj * 16, acc);
    }
    __syncthreads();

    // ---- Phase 3: Cw = Wc@c^T -> R1 (pitch 136); Sw = Ww@s^T -> R2; re-zero K-pad cols ----
    for (int idx = tid; idx < 112 * 16; idx += BT) {       // cols 112..127 (dirty from P layout)
        int r = idx >> 4, cp = 112 + (idx & 15);
        R1[r * PT + cp] = (bf16)0.f;
        R2[r * PT + cp] = (bf16)0.f;
    }
    for (int q = wid; q < 98; q += NW) {
        bool isCw = q < 49;
        int qq = isCw ? q : q - 49;
        int ti = qq / 7, tj = qq % 7;
        const bf16* Am = isCw ? WcS : WwS;
        const bf16* Bs = isCw ? cS : sS;
        v8f acc = {};
#pragma unroll
        for (int kc = 0; kc < 3; ++kc) {
            v16bf a  = ldA(Am, PD, ti * 16, kc * 32);
            v16bf bb = ldBt(Bs, PD, tj * 16, kc * 32);     // B = c^T / s^T
            acc = WMMA_BF16(a, bb, acc);
        }
        stD(isCw ? R1 : R2, PT, ti * 16, tj * 16, acc);
    }
    __syncthreads();

    // ---- Phase 4: Hc = tanh(Sw + Cw@F), Hw = tanh(Cw + Sw@F^T); whc/whw reductions ----
    for (int q = wid; q < 98; q += NW) {
        bool isHc = q < 49;
        int qq = isHc ? q : q - 49;
        int ti = qq / 7, tj = qq % 7;
        const bf16* Am = isHc ? R1 : R2;                   // Hc: A=Cw, Hw: A=Sw
        const bf16* Cm = isHc ? R2 : R1;                   // Hc: C=Sw, Hw: C=Cw
        const bf16* Bm = isHc ? FsT : Fs;                  // Hc: B=F via FsT, Hw: B=F^T via Fs
        v8f acc = ldC(Cm, PT, ti * 16, tj * 16);
#pragma unroll
        for (int kc = 0; kc < 4; ++kc) {                   // K = 128 (T/N padded)
            v16bf a  = ldA(Am, PT, ti * 16, kc * 32);
            v16bf bb = ldBt(Bm, PT, tj * 16, kc * 32);
            acc = WMMA_BF16(a, bb, acc);
        }
        const float* wv = isHc ? whcS : whwS;
        float* dst = isHc ? acS : awS;
        int col = lane & 15, rb = (lane >> 4) << 3;
        float partial = 0.f;
#pragma unroll
        for (int v = 0; v < 8; ++v) partial += wv[ti * 16 + rb + v] * TANH(acc[v]);
        atomicAdd(&dst[tj * 16 + col], partial);           // ds_add_f32
    }
    __syncthreads();

    // ---- Phase 5: softmax over 100 entries (wave 0 -> aw, wave 1 -> ac) ----
    if (wid < 2) {
        float* vec = wid ? acS : awS;
        float x[4]; float mx = -1e30f;
#pragma unroll
        for (int j = 0; j < 4; ++j) {
            int idx = lane + 32 * j;
            x[j] = (idx < 100) ? vec[idx] : -1e30f;
            mx = fmaxf(mx, x[j]);
        }
#pragma unroll
        for (int off = 16; off > 0; off >>= 1) mx = fmaxf(mx, __shfl_xor(mx, off, 32));
        float sum = 0.f;
#pragma unroll
        for (int j = 0; j < 4; ++j) {
            int idx = lane + 32 * j;
            float e = (idx < 100) ? __expf(x[j] - mx) : 0.f;
            x[j] = e; sum += e;
        }
#pragma unroll
        for (int off = 16; off > 0; off >>= 1) sum += __shfl_xor(sum, off, 32);
        float inv = 1.f / sum;
#pragma unroll
        for (int j = 0; j < 4; ++j) {
            int idx = lane + 32 * j;
            if (idx < 100) vec[idx] = x[j] * inv;
        }
    }
    __syncthreads();

    // ---- Phase 6: pooled outputs  co_w[l]=sum_n s[n,l]*Aw[n];  co_c[d]=sum_t c[t,d]*Ac[t] ----
    float* outp = Out + (size_t)b * 160;
    for (int o = tid; o < 160; o += BT) {
        float acc2 = 0.f;
        if (o < 80) {
            for (int n = 0; n < 100; ++n) acc2 += (float)sS[n * PD + o] * awS[n];
        } else {
            int d = o - 80;
            for (int t = 0; t < 100; ++t) acc2 += (float)cS[t * PD + d] * acS[t];
        }
        outp[o] = acc2;
    }
}

extern "C" void kernel_launch(void* const* d_in, const int* in_sizes, int n_in,
                              void* d_out, int out_size, void* d_ws, size_t ws_size,
                              hipStream_t stream) {
    const float* c_rep = (const float*)d_in[0];
    const float* s_rep = (const float*)d_in[1];
    const float* W_cw  = (const float*)d_in[2];
    const float* Wc    = (const float*)d_in[3];
    const float* Ww    = (const float*)d_in[4];
    const float* whw   = (const float*)d_in[5];
    const float* whc   = (const float*)d_in[6];
    float* out = (float*)d_out;

    (void)in_sizes; (void)n_in; (void)out_size; (void)d_ws; (void)ws_size;

    hipFuncSetAttribute((const void*)coattn_kernel,
                        hipFuncAttributeMaxDynamicSharedMemorySize, SMEM_BYTES);
    coattn_kernel<<<4096, BT, SMEM_BYTES, stream>>>(c_rep, s_rep, W_cw, Wc, Ww, whw, whc, out);
}
